// UNSATMinimizer_19490561590140
// MI455X (gfx1250) — matrix operand order repaired and checked by reading, not compiled, not run on hardware
//
#include <hip/hip_runtime.h>
#include <hip/hip_bf16.h>

#define DEV __device__ __forceinline__

constexpr int NV   = 100000;   // variables
constexpr int NC   = 420000;   // clauses
constexpr int NL   = 2 * NV;   // literals
constexpr int NNZE = 3 * NC;   // edges
constexpr int NROUNDS = 8;

typedef __attribute__((ext_vector_type(16))) _Float16 v16h;
typedef __attribute__((ext_vector_type(8)))  _Float16 v8h;
typedef __attribute__((ext_vector_type(4)))  _Float16 v4h;
typedef __attribute__((ext_vector_type(8)))  float    v8f;

// ---------------- small device helpers ----------------
DEV float softplus_f(float x) { return fmaxf(x, 0.f) + log1pf(__expf(-fabsf(x))); }
DEV float sigmoid_f(float x)  { return 1.f / (1.f + __expf(-x)); }

DEV unsigned pcg_hash(unsigned x) {
  x = x * 747796405u + 2891336453u;
  unsigned w = ((x >> ((x >> 28u) + 4u)) ^ x) * 277803737u;
  return (w >> 22u) ^ w;
}
DEV float u01(unsigned h) { return ((float)(h >> 8) + 0.5f) * (1.0f / 16777216.0f); }
DEV float gaussian(unsigned seed) {
  unsigned h1 = pcg_hash(seed);
  unsigned h2 = pcg_hash(h1 ^ 0x9E3779B9u);
  float u1 = u01(h1), u2 = u01(h2);
  return sqrtf(fmaxf(-2.f * __logf(u1), 0.f)) * __cosf(6.28318530718f * u2);
}

DEV void store4h(_Float16* p, float4 v) {
  *(v4h*)p = (v4h){(_Float16)v.x, (_Float16)v.y, (_Float16)v.z, (_Float16)v.w};
}

// ---------------- utility kernels ----------------
__global__ void k_fill(float* __restrict__ p, float v, long n) {
  long i = (long)blockIdx.x * blockDim.x + threadIdx.x;
  if (i < n) p[i] = v;
}

__global__ void k_edge_degree(const int* __restrict__ lit, float* __restrict__ cnt) {
  int e = blockIdx.x * blockDim.x + threadIdx.x;
  if (e < NNZE) atomicAdd(&cnt[lit[e]], 1.f);
}

__global__ void k_finalize_deg(const float* __restrict__ cnt,
                               float* __restrict__ dw, float* __restrict__ vdw) {
  int i = blockIdx.x * blockDim.x + threadIdx.x;
  if (i < NL) dw[i] = rsqrtf(fmaxf(cnt[i], 1.f));
  if (i < NV) vdw[i] = 4.f * rsqrtf(fmaxf(cnt[i] + cnt[i + NV], 1.f));
}

__global__ void k_graph_count(const int* __restrict__ gid, int n, float* __restrict__ cnt) {
  int i = blockIdx.x * blockDim.x + threadIdx.x;
  if (i < n) atomicAdd(&cnt[gid[i]], 1.f);
}

// ---------------- WMMA GEMM ----------------
// A-fragment loaders: pointer already offset to row*KP + kt*32 + halfSel*8.
DEV v16h loadA(const float* xk) {
  const float4 f0 = *(const float4*)(xk);
  const float4 f1 = *(const float4*)(xk + 4);
  const float4 f2 = *(const float4*)(xk + 16);
  const float4 f3 = *(const float4*)(xk + 20);
  v16h a;
  a[0] = (_Float16)f0.x; a[1] = (_Float16)f0.y; a[2] = (_Float16)f0.z; a[3] = (_Float16)f0.w;
  a[4] = (_Float16)f1.x; a[5] = (_Float16)f1.y; a[6] = (_Float16)f1.z; a[7] = (_Float16)f1.w;
  a[8] = (_Float16)f2.x; a[9] = (_Float16)f2.y; a[10] = (_Float16)f2.z; a[11] = (_Float16)f2.w;
  a[12] = (_Float16)f3.x; a[13] = (_Float16)f3.y; a[14] = (_Float16)f3.z; a[15] = (_Float16)f3.w;
  return a;
}
DEV v16h loadA(const _Float16* xk) {
  v8h lo = *(const v8h*)(xk);
  v8h hi = *(const v8h*)(xk + 16);
  v16h a;
#pragma unroll
  for (int i = 0; i < 8; ++i) { a[i] = lo[i]; a[8 + i] = hi[i]; }
  return a;
}
DEV void storeY(float* p, float v)    { *p = v; }
DEV void storeY(_Float16* p, float v) { *p = (_Float16)v; }

// Y(MxN) = act( X(MxKP) * W(KRxN) + b ). W f32 row-major, staged in LDS as f16
// pre-swizzled into B-fragment order: [kt][nt][lane][16 contiguous halves].
// Each wave computes MT row-tiles of 16 rows; block = 8 waves = 128*MT rows.
template<int KP, int KR, int N, int MT, bool RELU, typename TX, typename TY>
__launch_bounds__(256)
__global__ void gemm_wmma(const TX* __restrict__ X, const float* __restrict__ W,
                          const float* __restrict__ bias, TY* __restrict__ Y, int M) {
  constexpr int NT = N / 16;
  constexpr int KT = KP / 32;
  __shared__ __align__(16) _Float16 Wl[KP * N];
  const int tid = threadIdx.x;
  for (int idx = tid; idx < KP * N; idx += 256) {
    int i    = idx & 15;
    int lane = (idx >> 4) & 31;
    int t    = (idx >> 9) % NT;
    int kt   = idx / (512 * NT);
    int k = kt * 32 + (lane >> 4) * 8 + (i & 7) + (i >> 3) * 16;
    int n = t * 16 + (lane & 15);
    Wl[idx] = (k < KR) ? (_Float16)W[k * N + n] : (_Float16)0.0f;
  }
  __syncthreads();

  const int wave = tid >> 5, lane = tid & 31;
  const int rowBase = blockIdx.x * (128 * MT) + wave * (16 * MT);
  if (rowBase >= M) return;
  const int halfSel = lane >> 4;    // 0: lanes 0-15, 1: lanes 16-31
  const int l16 = lane & 15;

  bool valid[MT];
  const TX* xrow[MT];
#pragma unroll
  for (int m = 0; m < MT; ++m) {
    int rb = rowBase + m * 16;
    valid[m] = rb < M;
    xrow[m] = X + (long)((valid[m] ? rb : 0) + l16) * KP;
  }

  v8f acc[MT][NT];
#pragma unroll
  for (int t = 0; t < NT; ++t) {
    float b = bias[t * 16 + l16];
#pragma unroll
    for (int m = 0; m < MT; ++m) acc[m][t] = (v8f){b, b, b, b, b, b, b, b};
  }

#pragma unroll
  for (int kt = 0; kt < KT; ++kt) {
    v16h a[MT];
#pragma unroll
    for (int m = 0; m < MT; ++m) {
      const TX* xk = xrow[m] + kt * 32 + halfSel * 8;
      if (kt + 1 < KT) __builtin_prefetch(xk + 32, 0, 1);
      a[m] = loadA(xk);
    }
#pragma unroll
    for (int t = 0; t < NT; ++t) {
      const _Float16* bp = &Wl[((kt * NT + t) * 32 + lane) * 16];
      v8h blo = *(const v8h*)bp;
      v8h bhi = *(const v8h*)(bp + 8);
      v16h b;
#pragma unroll
      for (int i = 0; i < 8; ++i) { b[i] = blo[i]; b[8 + i] = bhi[i]; }
#pragma unroll
      for (int m = 0; m < MT; ++m)
        acc[m][t] = __builtin_amdgcn_wmma_f32_16x16x32_f16(
            false, a[m], false, b, (short)0, acc[m][t], false, false);
    }
  }

#pragma unroll
  for (int m = 0; m < MT; ++m) {
    if (!valid[m]) continue;
    int rb = rowBase + m * 16;
#pragma unroll
    for (int t = 0; t < NT; ++t) {
#pragma unroll
      for (int r = 0; r < 8; ++r) {
        int orow = rb + r + 8 * halfSel;
        float v = acc[m][t][r];
        if (RELU) v = fmaxf(v, 0.f);
        storeY(&Y[(long)orow * N + t * 16 + l16], v);
      }
    }
  }
}

// ---------------- round-phase kernels ----------------
// vq input (f16): [variables | noise4 | zero-pad to 96]
__global__ void k_build_vq_in(const float* __restrict__ var, _Float16* __restrict__ x1, int round) {
  long t = (long)blockIdx.x * blockDim.x + threadIdx.x;
  if (t >= (long)NV * 24) return;
  int v = (int)(t / 24);
  int f = (int)(t % 24) * 4;
  float4 out;
  if (f < 64) {
    out = *(const float4*)(var + (long)v * 64 + f);
  } else if (f == 64) {
    unsigned base = 0xA341316Cu ^ ((unsigned)(round + 1) * 0x9E3779B9u);
    out.x = gaussian(base + (unsigned)v * 4u + 0u);
    out.y = gaussian(base + (unsigned)v * 4u + 1u);
    out.z = gaussian(base + (unsigned)v * 4u + 2u);
    out.w = gaussian(base + (unsigned)v * 4u + 3u);
  } else {
    out = make_float4(0.f, 0.f, 0.f, 0.f);
  }
  store4h(x1 + (long)v * 96 + f, out);
}

// per-clause: cl[c] = exp(-sum_j softplus(sign*q[var_j])). warp per clause, 2 feats/lane.
__global__ void k_clause_forward(const int* __restrict__ lit, const float* __restrict__ q,
                                 float* __restrict__ cl) {
  long gt = (long)blockIdx.x * blockDim.x + threadIdx.x;
  int c = (int)(gt >> 5);
  int lane = threadIdx.x & 31;
  if (c >= NC) return;
  int l0 = lit[c * 3 + 0], l1 = lit[c * 3 + 1], l2 = lit[c * 3 + 2];
  int f = lane * 2;
  float cvx = 0.f, cvy = 0.f;
#pragma unroll
  for (int j = 0; j < 3; ++j) {
    int l = (j == 0) ? l0 : (j == 1) ? l1 : l2;
    float s = 1.f; int v = l;
    if (l >= NV) { s = -1.f; v = l - NV; }
    float2 qv = *(const float2*)(q + (long)v * 64 + f);
    cvx += softplus_f(s * qv.x);
    cvy += softplus_f(s * qv.y);
  }
  float2 out = make_float2(__expf(-cvx), __expf(-cvy));
  *(float2*)(cl + (long)c * 64 + f) = out;
}

// per (clause, 4-feat group): add scale*src[c][f..f+3] to dst[lit_j][f..f+3], 3 literals
__global__ void k_scatter3(const int* __restrict__ lit, const float* __restrict__ src,
                           int srcld, float scale, float* __restrict__ dst) {
  long t = (long)blockIdx.x * blockDim.x + threadIdx.x;
  if (t >= (long)NC * 16) return;
  int c = (int)(t >> 4);
  int f = (int)(t & 15) << 2;
  float4 s = *(const float4*)(src + (long)c * srcld + f);
  s.x *= scale; s.y *= scale; s.z *= scale; s.w *= scale;
#pragma unroll
  for (int j = 0; j < 3; ++j) {
    int l = lit[c * 3 + j];
    float* d = dst + (long)l * 64 + f;
    atomicAdd(d + 0, s.x); atomicAdd(d + 1, s.y);
    atomicAdd(d + 2, s.z); atomicAdd(d + 3, s.w);
  }
}

// variables_grad = vdw * (sig(q)*Gpos - sig(-q)*Gneg) -> ug_in cols 0..63; cols 64..127 = variables
__global__ void k_grad_pack(const float* __restrict__ q, const float* __restrict__ g,
                            const float* __restrict__ vdw, const float* __restrict__ var,
                            _Float16* __restrict__ x1) {
  long t = (long)blockIdx.x * blockDim.x + threadIdx.x;
  if (t >= (long)NV * 16) return;
  int v = (int)(t >> 4);
  int f = (int)(t & 15) << 2;
  float4 qv = *(const float4*)(q + (long)v * 64 + f);
  float4 gp = *(const float4*)(g + (long)v * 64 + f);
  float4 gn = *(const float4*)(g + (long)(v + NV) * 64 + f);
  float w = vdw[v];
  float4 o;
  o.x = w * (sigmoid_f(qv.x) * gp.x - sigmoid_f(-qv.x) * gn.x);
  o.y = w * (sigmoid_f(qv.y) * gp.y - sigmoid_f(-qv.y) * gn.y);
  o.z = w * (sigmoid_f(qv.z) * gp.z - sigmoid_f(-qv.z) * gn.z);
  o.w = w * (sigmoid_f(qv.w) * gp.w - sigmoid_f(-qv.w) * gn.w);
  store4h(x1 + (long)v * 256 + f, o);
  store4h(x1 + (long)v * 256 + 64 + f, *(const float4*)(var + (long)v * 64 + f));
}

// cm input (f16): [cscale*clauses | 4*cl]
__global__ void k_build_cm_in(const float* __restrict__ cls, const float* __restrict__ cl,
                              float cscale, _Float16* __restrict__ cmin) {
  long t = (long)blockIdx.x * blockDim.x + threadIdx.x;
  if (t >= (long)NC * 16) return;
  int c = (int)(t >> 4);
  int f = (int)(t & 15) << 2;
  float4 a = *(const float4*)(cls + (long)c * 64 + f);
  a.x *= cscale; a.y *= cscale; a.z *= cscale; a.w *= cscale;
  store4h(cmin + (long)c * 128 + f, a);
  float4 b = *(const float4*)(cl + (long)c * 64 + f);
  b.x *= 4.f; b.y *= 4.f; b.z *= 4.f; b.w *= 4.f;
  store4h(cmin + (long)c * 128 + 64 + f, b);
}

// ug_in cols 128..191 = dw[v]*VL[v], cols 192..255 = dw[v+NV]*VL[v+NV]
__global__ void k_pack_vl(const float* __restrict__ vl, const float* __restrict__ dw,
                          _Float16* __restrict__ x1) {
  long t = (long)blockIdx.x * blockDim.x + threadIdx.x;
  if (t >= (long)NV * 16) return;
  int v = (int)(t >> 4);
  int f = (int)(t & 15) << 2;
  float wp = dw[v], wn = dw[v + NV];
  float4 p = *(const float4*)(vl + (long)v * 64 + f);
  p.x *= wp; p.y *= wp; p.z *= wp; p.w *= wp;
  store4h(x1 + (long)v * 256 + 128 + f, p);
  float4 n = *(const float4*)(vl + (long)(v + NV) * 64 + f);
  n.x *= wn; n.y *= wn; n.z *= wn; n.w *= wn;
  store4h(x1 + (long)v * 256 + 192 + f, n);
}

// ---------------- PairNorm (3 passes, LDS-local segment accumulation) ----------------
__global__ void k_pn_sum(const float* __restrict__ x, int ldx, const int* __restrict__ gid,
                         int M, float* __restrict__ gsum) {
  __shared__ float ls[2048];
  for (int i = threadIdx.x; i < 2048; i += 256) ls[i] = 0.f;
  __syncthreads();
  int base = blockIdx.x * 1024;
  for (int i = threadIdx.x; i < 1024 * 32; i += 256) {
    int r = base + (i >> 5);
    if (r < M) {
      int f = (i & 31) * 2;
      int g = gid[r];
      float2 v2 = *(const float2*)(x + (long)r * ldx + f);
      atomicAdd(&ls[g * 64 + f], v2.x);
      atomicAdd(&ls[g * 64 + f + 1], v2.y);
    }
  }
  __syncthreads();
  for (int i = threadIdx.x; i < 2048; i += 256) {
    float s = ls[i];
    if (s != 0.f) atomicAdd(&gsum[i], s);
  }
}

__global__ void k_pn_center(float* __restrict__ x, int ldx, const int* __restrict__ gid, int M,
                            const float* __restrict__ gsum, const float* __restrict__ cnt,
                            float* __restrict__ gssq) {
  __shared__ float lssq[32];
  if (threadIdx.x < 32) lssq[threadIdx.x] = 0.f;
  __syncthreads();
  long gt = (long)blockIdx.x * blockDim.x + threadIdx.x;
  int r = (int)(gt >> 5);
  int lane = threadIdx.x & 31;
  float s = 0.f;
  int g = 0;
  bool act = r < M;
  if (act) {
    g = gid[r];
    float cn = fmaxf(cnt[g], 1.f);
    int f = lane * 2;
    float mx = gsum[g * 64 + f] / cn;
    float my = gsum[g * 64 + f + 1] / cn;
    float2 v2 = *(float2*)(x + (long)r * ldx + f);
    v2.x -= mx; v2.y -= my;
    *(float2*)(x + (long)r * ldx + f) = v2;
    s = v2.x * v2.x + v2.y * v2.y;
  }
#pragma unroll
  for (int o = 16; o > 0; o >>= 1) s += __shfl_xor(s, o, 32);
  if (act && lane == 0) atomicAdd(&lssq[g], s * (1.f / 64.f));
  __syncthreads();
  if (threadIdx.x < 32) {
    float v = lssq[threadIdx.x];
    if (v != 0.f) atomicAdd(&gssq[threadIdx.x], v);
  }
}

__global__ void k_pn_blend(const float* __restrict__ x, int ldx, const int* __restrict__ gid,
                           int M, const float* __restrict__ gssq, const float* __restrict__ cnt,
                           float* __restrict__ state, float blend) {
  long t = (long)blockIdx.x * blockDim.x + threadIdx.x;
  if (t >= (long)M * 16) return;
  int r = (int)(t >> 4);
  int f = (int)(t & 15) << 2;
  int g = gid[r];
  float cn = fmaxf(cnt[g], 1.f);
  float rs = rsqrtf(gssq[g] / cn + 1e-6f) * 0.25f;
  float4 xc = *(const float4*)(x + (long)r * ldx + f);
  float4 st = *(float4*)(state + (long)r * 64 + f);
  st.x = rs * xc.x + blend * st.x;
  st.y = rs * xc.y + blend * st.y;
  st.z = rs * xc.z + blend * st.z;
  st.w = rs * xc.w + blend * st.w;
  *(float4*)(state + (long)r * 64 + f) = st;
}

// ---------------- final readout: logit + noise -> sigmoid ----------------
__global__ void k_co_final(const float* __restrict__ h, const float* __restrict__ W1,
                           const float* __restrict__ b1, float* __restrict__ out) {
  long gt = (long)blockIdx.x * blockDim.x + threadIdx.x;
  int c = (int)(gt >> 5);
  int lane = threadIdx.x & 31;
  if (c >= NC) return;
  int f = lane * 2;
  float2 hv = *(const float2*)(h + (long)c * 64 + f);
  float2 wv = *(const float2*)(W1 + f);
  float s = hv.x * wv.x + hv.y * wv.y;
#pragma unroll
  for (int o = 16; o > 0; o >>= 1) s += __shfl_xor(s, o, 32);
  if (lane == 0) {
    float logit = s + b1[0] + gaussian(0xC0FFEEu ^ (unsigned)c);
    out[c] = sigmoid_f(logit);
  }
}

// ---------------- host orchestration ----------------
static inline int cdiv_l(long a, int b) { return (int)((a + b - 1) / b); }

extern "C" void kernel_launch(void* const* d_in, const int* in_sizes, int n_in,
                              void* d_out, int out_size, void* d_ws, size_t ws_size,
                              hipStream_t stream) {
  (void)in_sizes; (void)n_in; (void)out_size; (void)ws_size;
  const int* lit_idx    = (const int*)d_in[0];
  const int* var_gid    = (const int*)d_in[2];
  const int* clause_gid = (const int*)d_in[3];
  const float* vq_W0 = (const float*)d_in[4];  const float* vq_b0 = (const float*)d_in[5];
  const float* vq_W1 = (const float*)d_in[6];  const float* vq_b1 = (const float*)d_in[7];
  const float* cm_W0 = (const float*)d_in[8];  const float* cm_b0 = (const float*)d_in[9];
  const float* cm_W1 = (const float*)d_in[10]; const float* cm_b1 = (const float*)d_in[11];
  const float* ug_W0 = (const float*)d_in[12]; const float* ug_b0 = (const float*)d_in[13];
  const float* ug_W1 = (const float*)d_in[14]; const float* ug_b1 = (const float*)d_in[15];
  const float* ug_W2 = (const float*)d_in[16]; const float* ug_b2 = (const float*)d_in[17];
  const float* co_W0 = (const float*)d_in[18]; const float* co_b0 = (const float*)d_in[19];
  const float* co_W1 = (const float*)d_in[20]; const float* co_b1 = (const float*)d_in[21];
  float* out = (float*)d_out;

  char* base = (char*)d_ws;
  size_t off = 0;
  auto AF = [&](size_t nf) { float* p = (float*)(base + off); off = (off + nf * 4 + 15) & ~(size_t)15; return p; };
  auto AH = [&](size_t nh) { _Float16* p = (_Float16*)(base + off); off = (off + nh * 2 + 15) & ~(size_t)15; return p; };

  // f32 buffers
  float* B_var  = AF((size_t)NV * 64);   // variables state
  float* B_cls  = AF((size_t)NC * 64);   // clauses state (un-x0.2; consumers apply cscale)
  float* B_dw   = AF(NL);                // degree_weight
  float* B_vdw  = AF(NV);                // var_degree_weight
  float* B_cntv = AF(32);                // per-graph var counts
  float* B_cntc = AF(32);                // per-graph clause counts (contiguous with cntv)
  float* B_gsum = AF(2048);              // pair-norm feature sums
  float* B_gssq = AF(32);                // pair-norm variance sums (contiguous with gsum)
  float* B_q    = AF((size_t)NV * 64);   // query (f32); later ug output pre-norm
  float* B_cl   = AF((size_t)NC * 64);   // clause loss exp(-cv)
  float* B_g    = AF((size_t)NL * 64);   // literal grads; degree counts at init
  float* B_cd   = AF((size_t)NC * 128);  // clause_data (f32: scatter + pair-norm)
  // f16 buffers (GEMM-only intermediates)
  _Float16* H_x1  = AH((size_t)NV * 256);  // vq_in (stride 96) then ug_in (stride 256)
  _Float16* H_h   = AH((size_t)NV * 256);  // vq hidden (stride 64); later h1,h2 (stride 128)
  _Float16* H_cmi = AH((size_t)NC * 128);  // cm input
  _Float16* H_cmh = AH((size_t)NC * 128);  // cm hidden
  // aliases into dead regions
  _Float16* H_h1 = H_h;
  _Float16* H_h2 = H_h + (size_t)NV * 128;
  float* B_vl  = (float*)H_cmi;            // literal-loss scatter (L*64 f32 <= C*128 f16 bytes)
  float* B_coh = (float*)H_cmi;            // co hidden on final round (C*64 f32 == C*128 f16 bytes)

  const int T = 256;
  // one-time graph structure (recomputed every call; deterministic)
  k_fill<<<cdiv_l((long)NV * 64, T), T, 0, stream>>>(B_var, 1.f, (long)NV * 64);
  k_fill<<<cdiv_l((long)NC * 64, T), T, 0, stream>>>(B_cls, 1.f, (long)NC * 64);
  k_fill<<<cdiv_l(NL, T), T, 0, stream>>>(B_g, 0.f, NL);
  k_edge_degree<<<cdiv_l(NNZE, T), T, 0, stream>>>(lit_idx, B_g);
  k_finalize_deg<<<cdiv_l(NL, T), T, 0, stream>>>(B_g, B_dw, B_vdw);
  k_fill<<<1, 64, 0, stream>>>(B_cntv, 0.f, 64);
  k_graph_count<<<cdiv_l(NV, T), T, 0, stream>>>(var_gid, NV, B_cntv);
  k_graph_count<<<cdiv_l(NC, T), T, 0, stream>>>(clause_gid, NC, B_cntc);

  for (int round = 0; round < NROUNDS; ++round) {
    float cscale = (round == 0) ? 1.f : 0.2f;  // deferred end-of-round clause scaling
    // query MLP
    k_build_vq_in<<<cdiv_l((long)NV * 24, T), T, 0, stream>>>(B_var, H_x1, round);
    gemm_wmma<96, 68, 64, 2, true><<<cdiv_l(NV, 256), T, 0, stream>>>(H_x1, vq_W0, vq_b0, H_h, NV);
    gemm_wmma<64, 64, 64, 2, false><<<cdiv_l(NV, 256), T, 0, stream>>>(H_h, vq_W1, vq_b1, B_q, NV);
    // clause loss + gradient scatter
    k_clause_forward<<<cdiv_l((long)NC * 32, T), T, 0, stream>>>(lit_idx, B_q, B_cl);
    k_fill<<<cdiv_l((long)NL * 64, T), T, 0, stream>>>(B_g, 0.f, (long)NL * 64);
    k_scatter3<<<cdiv_l((long)NC * 16, T), T, 0, stream>>>(lit_idx, B_cl, 64, -1.f, B_g);
    k_grad_pack<<<cdiv_l((long)NV * 16, T), T, 0, stream>>>(B_q, B_g, B_vdw, B_var, H_x1);
    // clause MLP
    k_build_cm_in<<<cdiv_l((long)NC * 16, T), T, 0, stream>>>(B_cls, B_cl, cscale, H_cmi);
    gemm_wmma<128, 128, 128, 2, true><<<cdiv_l(NC, 256), T, 0, stream>>>(H_cmi, cm_W0, cm_b0, H_cmh, NC);
    gemm_wmma<128, 128, 128, 2, false><<<cdiv_l(NC, 256), T, 0, stream>>>(H_cmh, cm_W1, cm_b1, B_cd, NC);
    // variables_loss scatter + pack (B_vl aliases dead cm-input region)
    k_fill<<<cdiv_l((long)NL * 64, T), T, 0, stream>>>(B_vl, 0.f, (long)NL * 64);
    k_scatter3<<<cdiv_l((long)NC * 16, T), T, 0, stream>>>(lit_idx, B_cd, 128, 1.f, B_vl);
    k_pack_vl<<<cdiv_l((long)NV * 16, T), T, 0, stream>>>(B_vl, B_dw, H_x1);
    // clause pair-norm + blend
    k_fill<<<cdiv_l(2080, T), T, 0, stream>>>(B_gsum, 0.f, 2080);
    k_pn_sum<<<cdiv_l(NC, 1024), T, 0, stream>>>(B_cd + 64, 128, clause_gid, NC, B_gsum);
    k_pn_center<<<cdiv_l((long)NC * 32, T), T, 0, stream>>>(B_cd + 64, 128, clause_gid, NC,
                                                            B_gsum, B_cntc, B_gssq);
    k_pn_blend<<<cdiv_l((long)NC * 16, T), T, 0, stream>>>(B_cd + 64, 128, clause_gid, NC,
                                                           B_gssq, B_cntc, B_cls, 0.1f * cscale);
    // variable update MLP
    gemm_wmma<256, 256, 128, 2, true><<<cdiv_l(NV, 256), T, 0, stream>>>(H_x1, ug_W0, ug_b0, H_h1, NV);
    gemm_wmma<128, 128, 128, 2, true><<<cdiv_l(NV, 256), T, 0, stream>>>(H_h1, ug_W1, ug_b1, H_h2, NV);
    gemm_wmma<128, 128, 64, 2, false><<<cdiv_l(NV, 256), T, 0, stream>>>(H_h2, ug_W2, ug_b2, B_q, NV);
    // variable pair-norm + blend
    k_fill<<<cdiv_l(2080, T), T, 0, stream>>>(B_gsum, 0.f, 2080);
    k_pn_sum<<<cdiv_l(NV, 1024), T, 0, stream>>>(B_q, 64, var_gid, NV, B_gsum);
    k_pn_center<<<cdiv_l((long)NV * 32, T), T, 0, stream>>>(B_q, 64, var_gid, NV,
                                                            B_gsum, B_cntv, B_gssq);
    k_pn_blend<<<cdiv_l((long)NV * 16, T), T, 0, stream>>>(B_q, 64, var_gid, NV,
                                                           B_gssq, B_cntv, B_var, 0.1f);
    // readout only needed on the last round
    if (round == NROUNDS - 1) {
      gemm_wmma<64, 64, 64, 2, true><<<cdiv_l(NC, 256), T, 0, stream>>>(B_cls, co_W0, co_b0, B_coh, NC);
      k_co_final<<<cdiv_l((long)NC * 32, T), T, 0, stream>>>(B_coh, co_W1, co_b1, out);
    }
  }
}
